// ScaledDotProductAttention_16484084482486
// MI455X (gfx1250) — matrix operand order repaired and checked by reading, not compiled, not run on hardware
//
#include <hip/hip_runtime.h>

typedef __attribute__((ext_vector_type(16))) _Float16 v16h;
typedef __attribute__((ext_vector_type(8)))  float    v8f;
typedef __attribute__((ext_vector_type(4)))  int      v4i;

#if __has_builtin(__builtin_amdgcn_global_load_async_to_lds_b128) && \
    __has_builtin(__builtin_amdgcn_s_wait_asynccnt)
#define ASYNC_LDS 1
#else
#define ASYNC_LDS 0
#endif

namespace {
constexpr int   kB = 32;
constexpr int   kS = 2048;
constexpr int   kD = 64;
constexpr float kScale = 0.125f;              // 1 / sqrt(64)
constexpr int   kQTileRows     = 16;          // q rows per wave
constexpr int   kWaves         = 8;           // 256 threads (wave32)
constexpr int   kQRowsPerBlock = kQTileRows * kWaves;  // 128
constexpr int   kChunk         = 32;          // keys per staged chunk
constexpr int   kNChunks       = kS / kChunk; // 64
constexpr int   kLdsStride     = 68;          // padded row stride (floats): avoid bank conflicts
}

#define WMMA_F16(a, b, c) \
  __builtin_amdgcn_wmma_f32_16x16x32_f16(false, (a), false, (b), (short)0, (c), false, false)

// ---- global -> LDS, 16 bytes per lane --------------------------------------
__device__ __forceinline__ void copy16_g2l(float* l, const float* g) {
#if ASYNC_LDS
  // prototype (from clang diagnostic): (int4 __device__*, int4 __shared__*, imm offset, imm cpol)
  __builtin_amdgcn_global_load_async_to_lds_b128(
      (__attribute__((address_space(1))) v4i*)(g),
      (__attribute__((address_space(3))) v4i*)(l), 0, 0);
#else
  *reinterpret_cast<float4*>(l) = *reinterpret_cast<const float4*>(g);
#endif
}
__device__ __forceinline__ void wait_async_copies() {
#if ASYNC_LDS
  __builtin_amdgcn_s_wait_asynccnt(0);
#endif
}

// Stage a 32-row x 64-float chunk (row stride kD in global) into LDS
// (row stride kLdsStride). 512 x 16B transfers spread over 256 threads.
__device__ __forceinline__ void stage_chunk(float* dst, const float* src, int tid) {
#pragma unroll
  for (int r = 0; r < 2; ++r) {
    const int t   = tid + r * 256;  // 0..511
    const int row = t >> 4;
    const int seg = t & 15;
    copy16_g2l(dst + row * kLdsStride + seg * 4, src + row * kD + seg * 4);
  }
}

// A-layout (16x32 f16) from a global f32 row: half j -> K=kbase+j,
// one contiguous run of 16 in memory.
__device__ __forceinline__ v16h load_a_from_row(const float* row, int kbase) {
  v16h r;
  const float4 f0 = *reinterpret_cast<const float4*>(row + kbase);
  const float4 f1 = *reinterpret_cast<const float4*>(row + kbase + 4);
  const float4 f2 = *reinterpret_cast<const float4*>(row + kbase + 8);
  const float4 f3 = *reinterpret_cast<const float4*>(row + kbase + 12);
  r[0]  = (_Float16)f0.x; r[1]  = (_Float16)f0.y; r[2]  = (_Float16)f0.z; r[3]  = (_Float16)f0.w;
  r[4]  = (_Float16)f1.x; r[5]  = (_Float16)f1.y; r[6]  = (_Float16)f1.z; r[7]  = (_Float16)f1.w;
  r[8]  = (_Float16)f2.x; r[9]  = (_Float16)f2.y; r[10] = (_Float16)f2.z; r[11] = (_Float16)f2.w;
  r[12] = (_Float16)f3.x; r[13] = (_Float16)f3.y; r[14] = (_Float16)f3.z; r[15] = (_Float16)f3.w;
  return r;
}

// B-layout (32x16 f16): this lane's 16 K-values contiguous in (LDS) memory.
__device__ __forceinline__ v16h load_b_contig(const float* base) {
  v16h r;
#pragma unroll
  for (int q = 0; q < 4; ++q) {
    float4 f = reinterpret_cast<const float4*>(base)[q];
    r[4*q+0] = (_Float16)f.x; r[4*q+1] = (_Float16)f.y;
    r[4*q+2] = (_Float16)f.z; r[4*q+3] = (_Float16)f.w;
  }
  return r;
}

// B-layout for a V tile from LDS: 16 K-rows at stride kLdsStride, fixed column.
__device__ __forceinline__ v16h load_b_strided(const float* base) {
  v16h r;
#pragma unroll
  for (int h = 0; h < 16; ++h) r[h] = (_Float16)base[h * kLdsStride];
  return r;
}

// Per-lane online softmax update over this lane's score column.
__device__ __forceinline__ void online_update(const v8f& c, float* m, float* l) {
#pragma unroll
  for (int i = 0; i < 8; ++i) {
    const float s  = c[i] * kScale;
    const float mn = fmaxf(m[i], s);
    l[i] = l[i] * __expf(m[i] - mn) + __expf(s - mn);
    m[i] = mn;
  }
}

__global__ __launch_bounds__(256)
void sdpa_fwd_kernel(const float* __restrict__ q,
                     const float* __restrict__ kmat,
                     const float* __restrict__ v,
                     float* __restrict__ out,
                     float* __restrict__ attn) {
  __shared__ float    kbuf[2][kChunk * kLdsStride];
  __shared__ float    vbuf[2][kChunk * kLdsStride];
  __shared__ _Float16 plds[kWaves][kQTileRows * 32];

  const int tid  = threadIdx.x;
  const int lane = tid & 31;
  const int wave = tid >> 5;
  const int b    = blockIdx.y;
  const int q0   = blockIdx.x * kQRowsPerBlock + wave * kQTileRows;

  const int n      = lane & 15;   // N / column index (B and C layouts)
  const int hi     = lane >> 4;   // which 16-lane half
  const int kbaseA = hi ? 16 : 0;
  const int kbaseB = hi ? 16 : 0;
  const size_t bS  = (size_t)b * kS;

  // Q tile (16 rows) in A layout, split over D: a0 = d[0..31], a1 = d[32..63]
  const float* qrow = q + (bS + q0 + n) * kD;
  const v16h a0 = load_a_from_row(qrow,      kbaseA);
  const v16h a1 = load_a_from_row(qrow + 32, kbaseA);

  float mrow[8], lrow[8];
#pragma unroll
  for (int i = 0; i < 8; ++i) { mrow[i] = -3.0e38f; lrow[i] = 0.0f; }

  // ================= pass 1: per-lane online (m, l) over staged K chunks =====
  stage_chunk(kbuf[0], kmat + bS * kD, tid);
  for (int kc = 0; kc < kNChunks; ++kc) {
    wait_async_copies();
    __syncthreads();  // chunk kc resident; everyone done reading buf (kc+1)&1
    if (kc + 1 < kNChunks)
      stage_chunk(kbuf[(kc + 1) & 1], kmat + (bS + (size_t)(kc + 1) * kChunk) * kD, tid);

    const float* kb  = kbuf[kc & 1];
    const float* kr0 = kb + n * kLdsStride;           // sub-tile 0: keys k0+0..15
    const float* kr1 = kb + (16 + n) * kLdsStride;    // sub-tile 1: keys k0+16..31
    v16h b00 = load_b_contig(kr0 + kbaseB);
    v16h b01 = load_b_contig(kr0 + 32 + kbaseB);
    v16h b10 = load_b_contig(kr1 + kbaseB);
    v16h b11 = load_b_contig(kr1 + 32 + kbaseB);
    v8f c0 = {}, c1 = {};
    c0 = WMMA_F16(a0, b00, c0);
    c1 = WMMA_F16(a0, b10, c1);   // independent chain hides WMMA->WMMA hazard
    c0 = WMMA_F16(a1, b01, c0);
    c1 = WMMA_F16(a1, b11, c1);
    online_update(c0, mrow, lrow);
    online_update(c1, mrow, lrow);
  }

  // one-time merge of (m, l) across the 16-lane halves
#pragma unroll
  for (int step = 1; step < 16; step <<= 1) {
#pragma unroll
    for (int i = 0; i < 8; ++i) {
      const float mo = __shfl_xor(mrow[i], step, 32);
      const float lo = __shfl_xor(lrow[i], step, 32);
      const float mn = fmaxf(mrow[i], mo);
      lrow[i] = lrow[i] * __expf(mrow[i] - mn) + lo * __expf(mo - mn);
      mrow[i] = mn;
    }
  }
  float linv[8];
#pragma unroll
  for (int i = 0; i < 8; ++i) linv[i] = 1.0f / lrow[i];

  // ================= pass 2: recompute, write attn (NT), P @ V ==============
  __syncthreads();  // pass-1 LDS reads finished before restaging buf 0
  stage_chunk(kbuf[0], kmat + bS * kD, tid);
  stage_chunk(vbuf[0], v    + bS * kD, tid);

  v8f acc0 = {}, acc1 = {}, acc2 = {}, acc3 = {};
  _Float16* myLds = plds[wave];

  for (int kc = 0; kc < kNChunks; ++kc) {
    wait_async_copies();
    __syncthreads();
    if (kc + 1 < kNChunks) {
      const size_t nk = bS + (size_t)(kc + 1) * kChunk;
      stage_chunk(kbuf[(kc + 1) & 1], kmat + nk * kD, tid);
      stage_chunk(vbuf[(kc + 1) & 1], v    + nk * kD, tid);
    }

    const int    k0  = kc * kChunk;
    const float* kb  = kbuf[kc & 1];
    const float* kr0 = kb + n * kLdsStride;
    const float* kr1 = kb + (16 + n) * kLdsStride;
    v16h b00 = load_b_contig(kr0 + kbaseB);
    v16h b01 = load_b_contig(kr0 + 32 + kbaseB);
    v16h b10 = load_b_contig(kr1 + kbaseB);
    v16h b11 = load_b_contig(kr1 + 32 + kbaseB);
    v8f c0 = {}, c1 = {};
    c0 = WMMA_F16(a0, b00, c0);
    c1 = WMMA_F16(a0, b10, c1);
    c0 = WMMA_F16(a1, b01, c0);
    c1 = WMMA_F16(a1, b11, c1);

#pragma unroll
    for (int sub = 0; sub < 2; ++sub) {
      const v8f& c = sub ? c1 : c0;
#pragma unroll
      for (int i = 0; i < 8; ++i) {
        const int mr = i + hi * 8;
        const float p = __expf(c[i] * kScale - mrow[i]) * linv[i];
        // 512MB streaming output: non-temporal, keep K/V resident in cache
        __builtin_nontemporal_store(p, attn + (bS + q0 + mr) * (size_t)kS + k0 + sub * 16 + n);
        myLds[mr * 32 + sub * 16 + n] = (_Float16)p;
      }
    }

    // P tile (16x32) C-layout -> A-layout via wave-local LDS (DS is in-order)
    v16h ap;
    {
      const _Float16* base = myLds + n * 32;
#pragma unroll
      for (int j = 0; j < 8; ++j) ap[j]     = base[kbaseA + j];
#pragma unroll
      for (int j = 0; j < 8; ++j) ap[8 + j] = base[kbaseA + 8 + j];
    }

    // V chunk (32 x 64) as four 32x16 B tiles from LDS
    const float* vb = vbuf[kc & 1] + kbaseB * kLdsStride + n;
    acc0 = WMMA_F16(ap, load_b_strided(vb +  0), acc0);
    acc1 = WMMA_F16(ap, load_b_strided(vb + 16), acc1);
    acc2 = WMMA_F16(ap, load_b_strided(vb + 32), acc2);
    acc3 = WMMA_F16(ap, load_b_strided(vb + 48), acc3);
  }

  // ---- write output tile [16 x 64]
#pragma unroll
  for (int i = 0; i < 8; ++i) {
    const int mr = i + hi * 8;
    float* orow = out + (bS + q0 + mr) * kD + n;
    __builtin_nontemporal_store(acc0[i], orow + 0);
    __builtin_nontemporal_store(acc1[i], orow + 16);
    __builtin_nontemporal_store(acc2[i], orow + 32);
    __builtin_nontemporal_store(acc3[i], orow + 48);
  }
}

extern "C" void kernel_launch(void* const* d_in, const int* in_sizes, int n_in,
                              void* d_out, int out_size, void* d_ws, size_t ws_size,
                              hipStream_t stream) {
  (void)in_sizes; (void)n_in; (void)out_size; (void)d_ws; (void)ws_size;
  const float* q = (const float*)d_in[0];
  const float* k = (const float*)d_in[1];
  const float* v = (const float*)d_in[2];
  float* out  = (float*)d_out;
  float* attn = out + (size_t)kB * kS * kD;   // tuple output: (output, attn) flat

  dim3 grid(kS / kQRowsPerBlock, kB, 1);      // (16, 32)
  sdpa_fwd_kernel<<<grid, kWaves * 32, 0, stream>>>(q, k, v, out, attn);
}
